// Net_89343909691624
// MI455X (gfx1250) — matrix operand order repaired and checked by reading, not compiled, not run on hardware
//
#include <hip/hip_runtime.h>
#include <hip/hip_bf16.h>

typedef __attribute__((ext_vector_type(16))) _Float16 v16h;
typedef __attribute__((ext_vector_type(8)))  _Float16 h8;
typedef __attribute__((ext_vector_type(8)))  float    v8f;
typedef int v4i __attribute__((vector_size(16)));   // matches builtin's V4i

#define LDS_STRIDE 40   // halves; 80B row stride: 16B-aligned chunks, bank-friendly

#if __has_builtin(__builtin_amdgcn_global_load_async_to_lds_b128)
#define HAVE_ASYNC_LDS 1
#else
#define HAVE_ASYNC_LDS 0
#endif

// 16-byte global->LDS chunk copy: async (ASYNCcnt-tracked) when available.
__device__ __forceinline__ void copy16_g2l(const _Float16* g, _Float16* l) {
#if HAVE_ASYNC_LDS
    __builtin_amdgcn_global_load_async_to_lds_b128(
        (__attribute__((address_space(1))) v4i*)(g),
        (__attribute__((address_space(3))) v4i*)(l), 0, 0);
#else
    *(h8*)l = *(const h8*)g;
#endif
}
__device__ __forceinline__ void wait_async_copies() {
#if HAVE_ASYNC_LDS
 #if __has_builtin(__builtin_amdgcn_s_wait_asynccnt)
    __builtin_amdgcn_s_wait_asynccnt(0);
 #else
    asm volatile("s_wait_asynccnt 0x0" ::: "memory");
 #endif
#endif
}

// ---- fragment loaders (ISA 16x16x32 f16 layouts) --------------------------
__device__ __forceinline__ v16h load_frag_A(const _Float16* Al, int row, int hi) {
    // lane<16: K{0..7,16..23}; lane>=16: K{8..15,24..31}
    v16h a;
    ((h8*)&a)[0] = *(const h8*)&Al[row * LDS_STRIDE + hi * 8];
    ((h8*)&a)[1] = *(const h8*)&Al[row * LDS_STRIDE + 16 + hi * 8];
    return a;
}
__device__ __forceinline__ v16h load_frag_B(const _Float16* Bl, int col, int hi) {
    // lane<16: col=lane, K{0..7,16..23}; lane>=16: col=lane-16, K{8..15,24..31}
    v16h b;
    ((h8*)&b)[0] = *(const h8*)&Bl[col * LDS_STRIDE + hi * 8];
    ((h8*)&b)[1] = *(const h8*)&Bl[col * LDS_STRIDE + 16 + hi * 8];
    return b;
}
#define WMMA(a, b, c) __builtin_amdgcn_wmma_f32_16x16x32_f16(false, a, false, b, (short)0, c, false, false)

// ---------------------------------------------------------------------------
// Weight pre-conversion: f32 -> f16 (layout preserved)
// ---------------------------------------------------------------------------
__global__ __launch_bounds__(256)
void cvt_f16_kernel(const float* __restrict__ src, _Float16* __restrict__ dst,
                    long long total) {
    long long idx = (long long)blockIdx.x * 256 + threadIdx.x;
    if (idx < total) dst[idx] = (_Float16)src[idx];
}

// Weight pre-conversion with transpose: src [E][K][N] f32 -> dst [E][N][K] f16
__global__ __launch_bounds__(256)
void cvt_transpose_f16_kernel(const float* __restrict__ src, _Float16* __restrict__ dst,
                              int K, int N, long long total) {
    long long idx = (long long)blockIdx.x * 256 + threadIdx.x;
    if (idx >= total) return;
    long long kn = (long long)K * N;
    long long e  = idx / kn;
    long long r  = idx - e * kn;
    int k = (int)(r / N);
    int n = (int)(r - (long long)k * N);
    dst[e * kn + (long long)n * K + k] = (_Float16)src[idx];
}

// ---------------------------------------------------------------------------
// Conv1: direct fp32 (C=3 -> K=363, poor fit for K=32 matrix pipe; ~10% of FLOPs)
// ---------------------------------------------------------------------------
__global__ __launch_bounds__(256)
void conv1_relu_kernel(const float* __restrict__ X, const float* __restrict__ W,
                       const float* __restrict__ bias, float* __restrict__ Y) {
    int idx = blockIdx.x * 256 + threadIdx.x;
    const int total = 64 * 64 * 55 * 55;
    if (idx >= total) return;
    int ow = idx % 55; int t = idx / 55;
    int oh = t % 55;   t /= 55;
    int oc = t % 64;   int b = t / 64;
    float s = bias[oc];
    int ih0 = oh * 4 - 2, iw0 = ow * 4 - 2;
    const float* wbase = W + oc * 3 * 121;
    for (int ic = 0; ic < 3; ++ic) {
        const float* xb = X + ((b * 3 + ic) * 224) * 224;
        const float* wc = wbase + ic * 121;
        for (int kh = 0; kh < 11; ++kh) {
            int ih = ih0 + kh;
            if (ih < 0 || ih >= 224) continue;
            const float* xr = xb + ih * 224;
            const float* wr = wc + kh * 11;
            #pragma unroll
            for (int kw = 0; kw < 11; ++kw) {
                int iw = iw0 + kw;
                if (iw >= 0 && iw < 224) s += xr[iw] * wr[kw];
            }
        }
    }
    Y[idx] = s > 0.0f ? s : 0.0f;
}

// ---------------------------------------------------------------------------
// MaxPool 3x3 stride 2 VALID (window always in-bounds for 55/27/13 inputs)
// ---------------------------------------------------------------------------
__global__ __launch_bounds__(256)
void maxpool3s2_kernel(const float* __restrict__ X, float* __restrict__ Y,
                       int C, int H, int W, int OH, int OW, int total) {
    int idx = blockIdx.x * 256 + threadIdx.x;
    if (idx >= total) return;
    int ow = idx % OW; int t = idx / OW;
    int oh = t % OH;   t /= OH;
    int c  = t % C;    int b = t / C;
    const float* xb = X + ((b * C + c) * H + oh * 2) * W + ow * 2;
    float m = -3.4e38f;
    #pragma unroll
    for (int kh = 0; kh < 3; ++kh)
        #pragma unroll
        for (int kw = 0; kw < 3; ++kw) {
            float v = xb[kh * W + kw];
            m = v > m ? v : m;
        }
    Y[idx] = m;
}

// ---------------------------------------------------------------------------
// Implicit-GEMM conv (stride 1), f16 WMMA / f32 accum, fused bias+ReLU.
// Weights pre-converted to f16 [OC][K]. GEMM view: M=OC (mult 64),
// N=64*OH*OW, K=IC*KS*KS (mult 32). 256 threads = 8 waves; block tile
// 64(M) x 128(N); wave = 1 M-subtile x 4 N-subtiles -> 4 WMMA per stage
// sharing one A fragment. LDS double-buffered: one barrier per stage.
// A tile staged with async global->LDS 16B chunks.
// ---------------------------------------------------------------------------
template<int KS, int PAD>
__global__ __launch_bounds__(256)
void conv_igemm_wmma(const float* __restrict__ X, const _Float16* __restrict__ W16,
                     const float* __restrict__ bias, float* __restrict__ Y,
                     int IC, int IH, int IW, int OC, int OH, int OW) {
    constexpr int KS2 = KS * KS;
    const int K     = IC * KS2;
    const int Ntot  = 64 * OH * OW;
    const int nBase = blockIdx.x * 128;
    const int mBase = blockIdx.y * 64;
    const int t     = threadIdx.x;
    const int lane  = t & 31;
    const int wv    = t >> 5;
    const int wm    = (wv & 3) * 16;        // wave M-subtile row base
    const int wn    = (wv >> 2) * 64;       // wave N-group base (0 or 64)

    __shared__ __align__(16) _Float16 lA[2][64 * LDS_STRIDE];
    __shared__ __align__(16) _Float16 lB[2][128 * LDS_STRIDE];

    // A staging: one 16B chunk per thread (64 rows x 4 chunks)
    const int am = t >> 2;                  // 0..63
    const int ak = (t & 3) * 8;             // 0,8,16,24
    const _Float16* wsrc = W16 + (long long)(mBase + am) * K + ak;

    // B staging: one im2col column, 16 consecutive K values
    const int bcol = t & 127;
    const int bkh  = (t >> 7) * 16;
    int n = nBase + bcol; if (n >= Ntot) n = Ntot - 1;   // clamp (never stored)
    const int ow_ = n % OW;
    int tn = n / OW;
    const int oh_ = tn % OH;
    const int b_  = tn / OH;
    const float* xb = X + (long long)b_ * IC * IH * IW;

    v8f acc0 = {}, acc1 = {}, acc2 = {}, acc3 = {};

    const int nstage = K >> 5;
    for (int s = 0; s < nstage; ++s) {
        const int k0 = s << 5;
        _Float16* A  = &lA[s & 1][0];
        _Float16* Bl = &lB[s & 1][0];

        // stage A (f16 weights, contiguous 16B chunk)
        copy16_g2l(wsrc + k0, &A[am * LDS_STRIDE + ak]);

        // stage B (im2col gather; constant-divisor decompose once per stage)
        {
            int k  = k0 + bkh;
            int ic = k / KS2;
            int r  = k - ic * KS2;
            int kr = r / KS;
            int kc = r - kr * KS;
            #pragma unroll
            for (int i = 0; i < 16; ++i) {
                int ih = oh_ + kr - PAD, iw = ow_ + kc - PAD;
                float v = 0.0f;
                if (ih >= 0 && ih < IH && iw >= 0 && iw < IW)
                    v = xb[((long long)ic * IH + ih) * IW + iw];
                Bl[bcol * LDS_STRIDE + bkh + i] = (_Float16)v;
                if (++kc == KS) { kc = 0; if (++kr == KS) { kr = 0; ++ic; } }
            }
        }
        wait_async_copies();
        __syncthreads();   // one barrier per stage (double-buffered LDS)

        const int hi = lane >> 4, mrow = lane & 15;
        v16h a  = load_frag_A(A, wm + mrow, hi);
        v16h b0 = load_frag_B(Bl, wn + 0  + mrow, hi);
        v16h b1 = load_frag_B(Bl, wn + 16 + mrow, hi);
        v16h b2 = load_frag_B(Bl, wn + 32 + mrow, hi);
        v16h b3 = load_frag_B(Bl, wn + 48 + mrow, hi);
        acc0 = WMMA(a, b0, acc0);
        acc1 = WMMA(a, b1, acc1);
        acc2 = WMMA(a, b2, acc2);
        acc3 = WMMA(a, b3, acc3);
    }

    // store: D VGPR r -> (M = r + (lane>=16)*8, N = lane&15)
    const int hi = lane >> 4, nc = lane & 15;
    v8f accs[4] = {acc0, acc1, acc2, acc3};
    #pragma unroll
    for (int j = 0; j < 4; ++j) {
        int no = nBase + wn + j * 16 + nc;
        if (no < Ntot) {
            int ow2 = no % OW;
            int t2  = no / OW;
            int oh2 = t2 % OH;
            int b2  = t2 / OH;
            #pragma unroll
            for (int r = 0; r < 8; ++r) {
                int oc  = mBase + wm + hi * 8 + r;
                float v = accs[j][r] + bias[oc];
                v = v > 0.0f ? v : 0.0f;
                Y[((long long)(b2 * OC + oc) * OH + oh2) * OW + ow2] = v;
            }
        }
    }
}

// ---------------------------------------------------------------------------
// Batched GEMM: C[e] = act(A[e][64,K] @ Bt[e]^T + bias[e][N]), f16 WMMA.
// A: f32 [64,K] row-major. Bt: f16 [N][K] (pre-transposed weights) -> B tile
// columns are contiguous 16B chunks, staged with async global->LDS copies,
// OOB columns clamped (never stored). 256 threads = 8 waves; block tile
// 64(M, whole batch) x 64(N); wave = 1 M-subtile x 2 N-subtiles.
// Runs K/32 full stages; callers pass 9216, 288, 144 (144 % 32 = 16: last 16 K skipped).
// ---------------------------------------------------------------------------
__global__ __launch_bounds__(256)
void gemm_bias_act_wmma(const float* __restrict__ A, const _Float16* __restrict__ Bt,
                        const float* __restrict__ bias, float* __restrict__ C,
                        int N, int K,
                        long long sA, long long sB, long long sBias, long long sC,
                        int act) {
    const int e = blockIdx.z;
    const float*    Ae = A    + (long long)e * sA;
    const _Float16* Be = Bt   + (long long)e * sB;
    const float*    be = bias + (long long)e * sBias;
    float*          Ce = C    + (long long)e * sC;

    const int nBase = blockIdx.x * 64;
    const int t     = threadIdx.x;
    const int lane  = t & 31;
    const int wv    = t >> 5;
    const int wm    = (wv & 3) * 16;
    const int wn    = (wv >> 2) * 32;

    __shared__ __align__(16) _Float16 lA[2][64 * LDS_STRIDE];
    __shared__ __align__(16) _Float16 lB[2][64 * LDS_STRIDE];

    // A staging: 8 contiguous f32 per thread, cvt -> f16
    const int am = t >> 2;                  // 0..63 (batch row)
    const int ak = (t & 3) * 8;             // 0,8,16,24
    const float* asrc = Ae + (long long)am * K + ak;

    // B staging: one 16B chunk per thread from transposed f16 weights
    const int bcol   = t & 63;
    const int bchunk = (t >> 6) * 8;        // 0,8,16,24
    int bn = nBase + bcol; if (bn >= N) bn = N - 1;   // clamp (never stored)
    const _Float16* bsrc = Be + (long long)bn * K + bchunk;

    v8f acc0 = {}, acc1 = {};

    const int nstage = K >> 5;
    for (int s = 0; s < nstage; ++s) {
        const int k0 = s << 5;
        _Float16* Al = &lA[s & 1][0];
        _Float16* Bl = &lB[s & 1][0];
        {
            const float* src = asrc + k0;
            #pragma unroll
            for (int i = 0; i < 8; ++i)
                Al[am * LDS_STRIDE + ak + i] = (_Float16)src[i];
        }
        copy16_g2l(bsrc + k0, &Bl[bcol * LDS_STRIDE + bchunk]);

        wait_async_copies();
        __syncthreads();

        const int hi = lane >> 4, mrow = lane & 15;
        v16h a  = load_frag_A(Al, wm + mrow, hi);
        v16h b0 = load_frag_B(Bl, wn + 0  + mrow, hi);
        v16h b1 = load_frag_B(Bl, wn + 16 + mrow, hi);
        acc0 = WMMA(a, b0, acc0);
        acc1 = WMMA(a, b1, acc1);
    }

    const int hi = lane >> 4, nc = lane & 15;
    v8f accs[2] = {acc0, acc1};
    #pragma unroll
    for (int j = 0; j < 2; ++j) {
        int no = nBase + wn + j * 16 + nc;
        if (no < N) {
            float bv = be[no];
            #pragma unroll
            for (int r = 0; r < 8; ++r) {
                int row = wm + hi * 8 + r;
                float v = accs[j][r] + bv;
                if (act) v = v > 0.0f ? v : 0.0f;
                Ce[(long long)row * N + no] = v;
            }
        }
    }
}

// ---------------------------------------------------------------------------
// Gate: logits = g @ gw2 + gb2 ; top-2 ; softmax over top-2 -> gates[64,8]
// ---------------------------------------------------------------------------
__global__ __launch_bounds__(64)
void gate_topk_kernel(const float* __restrict__ g, const float* __restrict__ gw2,
                      const float* __restrict__ gb2, float* __restrict__ gates) {
    int b = threadIdx.x;
    if (b >= 64) return;
    float logit[8];
    #pragma unroll
    for (int e = 0; e < 8; ++e) {
        float s = gb2[e];
        for (int j = 0; j < 72; ++j) s += g[b * 72 + j] * gw2[j * 8 + e];
        logit[e] = s;
    }
    int i0 = 0;
    #pragma unroll
    for (int e = 1; e < 8; ++e) if (logit[e] > logit[i0]) i0 = e;
    int i1 = (i0 == 0) ? 1 : 0;
    #pragma unroll
    for (int e = 0; e < 8; ++e)
        if (e != i0 && logit[e] > logit[i1]) i1 = e;
    float w1  = __expf(logit[i1] - logit[i0]);   // logit[i0] is the max
    float inv = 1.0f / (1.0f + w1);
    #pragma unroll
    for (int e = 0; e < 8; ++e) gates[b * 8 + e] = 0.0f;
    gates[b * 8 + i0] = inv;
    gates[b * 8 + i1] = w1 * inv;
}

// ---------------------------------------------------------------------------
// Combine: out[b,o] = sum_e gates[b,e] * eo[e,b,o]
// ---------------------------------------------------------------------------
__global__ __launch_bounds__(256)
void combine_kernel(const float* __restrict__ eo, const float* __restrict__ gates,
                    float* __restrict__ out) {
    int idx = blockIdx.x * 256 + threadIdx.x;
    if (idx >= 64 * 1000) return;
    int o = idx % 1000, b = idx / 1000;
    float s = 0.0f;
    #pragma unroll
    for (int e = 0; e < 8; ++e)
        s += gates[b * 8 + e] * eo[((long long)e * 64 + b) * 1000 + o];
    out[idx] = s;
}

// ---------------------------------------------------------------------------
// Host-side orchestration
// ---------------------------------------------------------------------------
extern "C" void kernel_launch(void* const* d_in, const int* in_sizes, int n_in,
                              void* d_out, int out_size, void* d_ws, size_t ws_size,
                              hipStream_t stream) {
    const float* x   = (const float*)d_in[0];
    const float* c1w = (const float*)d_in[1];  const float* c1b = (const float*)d_in[2];
    const float* c2w = (const float*)d_in[3];  const float* c2b = (const float*)d_in[4];
    const float* c3w = (const float*)d_in[5];  const float* c3b = (const float*)d_in[6];
    const float* c4w = (const float*)d_in[7];  const float* c4b = (const float*)d_in[8];
    const float* c5w = (const float*)d_in[9];  const float* c5b = (const float*)d_in[10];
    const float* gw1 = (const float*)d_in[11]; const float* gb1 = (const float*)d_in[12];
    const float* gw2 = (const float*)d_in[13]; const float* gb2 = (const float*)d_in[14];
    const float* ew1 = (const float*)d_in[15]; const float* eb1 = (const float*)d_in[16];
    const float* ew2 = (const float*)d_in[17]; const float* eb2 = (const float*)d_in[18];
    const float* ew3 = (const float*)d_in[19]; const float* eb3 = (const float*)d_in[20];

    // Workspace layout (bytes); conv activations ping-pong between A and B.
    char* ws = (char*)d_ws;
    const size_t OFF_A     = 0;                          // max 12,390,400 f (conv1 out)
    const size_t OFF_B     = 49561600;                   // max  2,985,984 f (pool1)
    const size_t OFF_FEAT  = OFF_B     + 11943936;       // 589,824 f
    const size_t OFF_G1    = OFF_FEAT  + 2359296;        // 4,608 f
    const size_t OFF_GATES = OFF_G1    + 18432;          // 512 f
    const size_t OFF_H1    = OFF_GATES + 2048;           // 147,456 f
    const size_t OFF_H2    = OFF_H1    + 589824;         // 73,728 f
    const size_t OFF_EO    = OFF_H2    + 294912;         // 512,000 f
    // f16 weight caches
    const size_t OFF_C2W   = OFF_EO    + 2048000;        // 307,200 h
    const size_t OFF_C3W   = OFF_C2W   + 614400;         // 663,552 h
    const size_t OFF_C4W   = OFF_C3W   + 1327104;        // 884,736 h
    const size_t OFF_C5W   = OFF_C4W   + 1769472;        // 589,824 h
    const size_t OFF_GW1T  = OFF_C5W   + 1179648;        // 663,552 h   [72][9216]
    const size_t OFF_EW1T  = OFF_GW1T  + 1327104;        // 21,233,664 h [8][288][9216]
    const size_t OFF_EW2T  = OFF_EW1T  + 42467328;       // 331,776 h   [8][144][288]
    const size_t OFF_EW3T  = OFF_EW2T  + 663552;         // 1,152,000 h [8][1000][144]
    float*     bufA  = (float*)(ws + OFF_A);
    float*     bufB  = (float*)(ws + OFF_B);
    float*     feat  = (float*)(ws + OFF_FEAT);
    float*     g1    = (float*)(ws + OFF_G1);
    float*     gates = (float*)(ws + OFF_GATES);
    float*     h1    = (float*)(ws + OFF_H1);
    float*     h2    = (float*)(ws + OFF_H2);
    float*     eo    = (float*)(ws + OFF_EO);
    _Float16*  c2w16 = (_Float16*)(ws + OFF_C2W);
    _Float16*  c3w16 = (_Float16*)(ws + OFF_C3W);
    _Float16*  c4w16 = (_Float16*)(ws + OFF_C4W);
    _Float16*  c5w16 = (_Float16*)(ws + OFF_C5W);
    _Float16*  gw1t  = (_Float16*)(ws + OFF_GW1T);
    _Float16*  ew1t  = (_Float16*)(ws + OFF_EW1T);
    _Float16*  ew2t  = (_Float16*)(ws + OFF_EW2T);
    _Float16*  ew3t  = (_Float16*)(ws + OFF_EW3T);

    // ---- weight pre-conversion (f32 -> f16; GEMM weights also transposed)
    {
        long long n;
        n = 307200;   cvt_f16_kernel<<<(int)((n+255)/256),256,0,stream>>>(c2w, c2w16, n);
        n = 663552;   cvt_f16_kernel<<<(int)((n+255)/256),256,0,stream>>>(c3w, c3w16, n);
        n = 884736;   cvt_f16_kernel<<<(int)((n+255)/256),256,0,stream>>>(c4w, c4w16, n);
        n = 589824;   cvt_f16_kernel<<<(int)((n+255)/256),256,0,stream>>>(c5w, c5w16, n);
        n = 663552;   cvt_transpose_f16_kernel<<<(int)((n+255)/256),256,0,stream>>>(gw1, gw1t, 9216, 72,  n);
        n = 21233664; cvt_transpose_f16_kernel<<<(int)((n+255)/256),256,0,stream>>>(ew1, ew1t, 9216, 288, n);
        n = 331776;   cvt_transpose_f16_kernel<<<(int)((n+255)/256),256,0,stream>>>(ew2, ew2t, 288,  144, n);
        n = 1152000;  cvt_transpose_f16_kernel<<<(int)((n+255)/256),256,0,stream>>>(ew3, ew3t, 144,  1000, n);
    }

    // ---- conv1 + ReLU : x -> bufA [64,64,55,55]
    {
        int total = 64 * 64 * 55 * 55;
        conv1_relu_kernel<<<(total + 255) / 256, 256, 0, stream>>>(x, c1w, c1b, bufA);
    }
    // ---- pool1 : bufA -> bufB [64,64,27,27]
    {
        int total = 64 * 64 * 27 * 27;
        maxpool3s2_kernel<<<(total + 255) / 256, 256, 0, stream>>>(
            bufA, bufB, 64, 55, 55, 27, 27, total);
    }
    // ---- conv2 + ReLU (WMMA igemm) : bufB -> bufA [64,192,27,27]  N=46656 K=1600
    conv_igemm_wmma<5, 2><<<dim3((46656 + 127) / 128, 192 / 64), 256, 0, stream>>>(
        bufB, c2w16, c2b, bufA, 64, 27, 27, 192, 27, 27);
    // ---- pool2 : bufA -> bufB [64,192,13,13]
    {
        int total = 64 * 192 * 13 * 13;
        maxpool3s2_kernel<<<(total + 255) / 256, 256, 0, stream>>>(
            bufA, bufB, 192, 27, 27, 13, 13, total);
    }
    // ---- conv3 + ReLU : bufB -> bufA [64,384,13,13]  N=10816 K=1728
    conv_igemm_wmma<3, 1><<<dim3((10816 + 127) / 128, 384 / 64), 256, 0, stream>>>(
        bufB, c3w16, c3b, bufA, 192, 13, 13, 384, 13, 13);
    // ---- conv4 + ReLU : bufA -> bufB [64,256,13,13]  K=3456
    conv_igemm_wmma<3, 1><<<dim3((10816 + 127) / 128, 256 / 64), 256, 0, stream>>>(
        bufA, c4w16, c4b, bufB, 384, 13, 13, 256, 13, 13);
    // ---- conv5 + ReLU : bufB -> bufA [64,256,13,13]  K=2304
    conv_igemm_wmma<3, 1><<<dim3((10816 + 127) / 128, 256 / 64), 256, 0, stream>>>(
        bufB, c5w16, c5b, bufA, 256, 13, 13, 256, 13, 13);
    // ---- pool5 : bufA -> feat [64,256,6,6] == [64,9216]
    {
        int total = 64 * 256 * 6 * 6;
        maxpool3s2_kernel<<<(total + 255) / 256, 256, 0, stream>>>(
            bufA, feat, 256, 13, 13, 6, 6, total);
    }
    // ---- gate layer 1 : g1 = relu(feat @ gw1 + gb1)   [64,72]
    gemm_bias_act_wmma<<<dim3(2, 1, 1), 256, 0, stream>>>(
        feat, gw1t, gb1, g1, 72, 9216, 0LL, 0LL, 0LL, 0LL, 1);
    // ---- gate top-2 softmax -> gates [64,8]
    gate_topk_kernel<<<1, 64, 0, stream>>>(g1, gw2, gb2, gates);
    // ---- expert layer 1 : h1 = relu(feat @ ew1[e] + eb1[e])   [8,64,288]
    gemm_bias_act_wmma<<<dim3(5, 1, 8), 256, 0, stream>>>(
        feat, ew1t, eb1, h1, 288, 9216,
        0LL, (long long)288 * 9216, 288LL, (long long)64 * 288, 1);
    // ---- expert layer 2 : h2 = relu(h1 @ ew2[e] + eb2[e])     [8,64,144]
    gemm_bias_act_wmma<<<dim3(3, 1, 8), 256, 0, stream>>>(
        h1, ew2t, eb2, h2, 144, 288,
        (long long)64 * 288, (long long)144 * 288, 144LL, (long long)64 * 144, 1);
    // ---- expert layer 3 : eo = h2 @ ew3[e] + eb3[e]           [8,64,1000]
    gemm_bias_act_wmma<<<dim3(16, 1, 8), 256, 0, stream>>>(
        h2, ew3t, eb3, eo, 1000, 144,
        (long long)64 * 144, (long long)1000 * 144, 1000LL, (long long)64 * 1000, 0);
    // ---- gated combine -> d_out [64,1000]
    combine_kernel<<<(64000 + 255) / 256, 256, 0, stream>>>(eo, gates, (float*)d_out);
}
